// GCN_16638703305297
// MI455X (gfx1250) — matrix operand order repaired and checked by reading, not compiled, not run on hardware
//
#include <hip/hip_runtime.h>
#include <hip/hip_bf16.h>

#define N_NODES 100000
#define N_EDGES 1600000
#define HIDDEN  128
#define N_STRUCT 6
#define EMB_DIM 122
#define BN_EPS 1e-5f

typedef __attribute__((ext_vector_type(16))) __bf16 v16bf;
typedef __attribute__((ext_vector_type(8)))  float  v8f;

static __device__ __forceinline__ __bf16 f2bf(float f) {
    return (__bf16)f;   // RNE f32->bf16 convert (v_cvt path on gfx1250)
}

// ---------------------------------------------------------------- utilities
__global__ void gcn_zero_f4(float4* __restrict__ p, int n4) {
    int i = blockIdx.x * blockDim.x + threadIdx.x;
    if (i < n4) p[i] = make_float4(0.f, 0.f, 0.f, 0.f);
}

__global__ void gcn_deg(const int* __restrict__ dst, float* __restrict__ deg) {
    int e = blockIdx.x * blockDim.x + threadIdx.x;
    if (e < N_EDGES) atomicAdd(deg + dst[e], 1.0f);
}

__global__ void gcn_dinv(float* __restrict__ deg) {
    int n = blockIdx.x * blockDim.x + threadIdx.x;
    if (n < N_NODES) {
        float d = deg[n];
        deg[n] = (d > 0.f) ? rsqrtf(d) : 0.f;
    }
}

// x = [emb || relu(struct @ proj_w + proj_b)]
__global__ void gcn_feat(const float* __restrict__ emb,
                         const float* __restrict__ sf,
                         const float* __restrict__ pw,
                         const float* __restrict__ pb,
                         float* __restrict__ x) {
    int idx = blockIdx.x * blockDim.x + threadIdx.x;
    if (idx >= N_NODES * HIDDEN) return;
    int n = idx >> 7;
    int c = idx & 127;
    float v;
    if (c < EMB_DIM) {
        v = emb[(size_t)n * EMB_DIM + c];
    } else {
        int j = c - EMB_DIM;
        float s = pb[j];
#pragma unroll
        for (int k = 0; k < N_STRUCT; ++k)
            s += sf[n * N_STRUCT + k] * pw[k * N_STRUCT + j];
        v = fmaxf(s, 0.f);
    }
    x[idx] = v;
}

// ------------------------------------------------------- WMMA GEMM  h = x@W
// One block = 256 threads = 8 waves. Each wave computes a 16x128 output strip
// using v_wmma_f32_16x16x32_bf16 (bf16 A/B, fp32 accumulate over K=128).
// W (128x128 f32, row-major [k][n]) is staged to LDS transposed [n][k] as
// bf16, stride 136 (272 B: 16B-aligned, bank step 4 -> conflict-free).
#define SW_STRIDE 136

__global__ __launch_bounds__(256) void gcn_gemm_wmma(
        const float* __restrict__ x, const float* __restrict__ W,
        float* __restrict__ h, int n_tiles) {
    __shared__ __bf16 sW[HIDDEN * SW_STRIDE];

    for (int idx = threadIdx.x; idx < HIDDEN * HIDDEN; idx += 256) {
        int k = idx >> 7;
        int n = idx & 127;
        sW[n * SW_STRIDE + k] = f2bf(W[idx]);
    }
    __syncthreads();

    const int wave = threadIdx.x >> 5;
    const int lane = threadIdx.x & 31;
    const int tile = blockIdx.x * 8 + wave;
    if (tile >= n_tiles) return;           // wave-uniform: EXEC stays all-1s

    const int half = lane >> 4;            // K-half selector per ISA layout
    const int mn   = lane & 15;            // row (A) or col (B/D) id

    const float* xrow = x + (size_t)(tile * 16 + mn) * HIDDEN;

    v8f acc[8] = {};

#pragma unroll
    for (int ks = 0; ks < 4; ++ks) {
        const int kbase = ks * 32;
        // A tile 16x32 bf16: reg i holds K pair; lanes 16-31 take K+8 / K+24
        v16bf a;
#pragma unroll
        for (int i = 0; i < 8; ++i) {
            int k0 = kbase + ((i >> 2) << 4) + (half << 3) + ((i & 3) << 1);
            a[2 * i]     = f2bf(xrow[k0]);
            a[2 * i + 1] = f2bf(xrow[k0 + 1]);
        }
#pragma unroll
        for (int nt = 0; nt < 8; ++nt) {
            // B tile 32x16 bf16: lanes 0-15 K=0..7,16..23, lanes 16-31 K=8..15,24..31
            const __bf16* wp = sW + (nt * 16 + mn) * SW_STRIDE + kbase + (half << 3);
            v16bf b;
#pragma unroll
            for (int i = 0; i < 16; ++i) b[i] = wp[(i & 7) + ((i >> 3) << 4)];
            acc[nt] = __builtin_amdgcn_wmma_f32_16x16x32_bf16(
                false, a, false, b, (short)0, acc[nt], false, false);
        }
    }

    // D layout: VGPR r -> M=r (lanes 0-15), M=r+8 (lanes 16-31); N=mn
#pragma unroll
    for (int nt = 0; nt < 8; ++nt) {
        float* hp = h + (size_t)tile * 16 * HIDDEN + nt * 16 + mn;
#pragma unroll
        for (int r = 0; r < 8; ++r) {
            int m = r + (half << 3);
            hp[(size_t)m * HIDDEN] = acc[nt][r];
        }
    }
}

// ----------------------------------------------- edge gather + scatter-add
// One wave per edge: lanes share src/dst, each lane moves one float4 of the
// 128-wide row -> coalesced 512B gather + global_atomic_add_f32 scatter.
__global__ void gcn_scatter(const int* __restrict__ src,
                            const int* __restrict__ dst,
                            const float* __restrict__ dinv,
                            const float* __restrict__ h,
                            float* __restrict__ agg) {
    long long t = (long long)blockIdx.x * blockDim.x + threadIdx.x;
    int e = (int)(t >> 5);
    if (e >= N_EDGES) return;
    int c = (int)(t & 31) << 2;
    int s = src[e], d = dst[e];
    float nrm = dinv[s] * dinv[d];
    const float4 hv = *(const float4*)(h + (size_t)s * HIDDEN + c);
    float* out = agg + (size_t)d * HIDDEN + c;
    atomicAdd(out + 0, hv.x * nrm);
    atomicAdd(out + 1, hv.y * nrm);
    atomicAdd(out + 2, hv.z * nrm);
    atomicAdd(out + 3, hv.w * nrm);
}

// ------------------------------------------ bias + BN(eval) + ReLU + resid
__global__ void gcn_bn_relu(const float* __restrict__ agg,
                            const float* __restrict__ xprev,
                            const float* __restrict__ bias,
                            const float* __restrict__ gamma,
                            const float* __restrict__ beta,
                            const float* __restrict__ mean,
                            const float* __restrict__ var,
                            float* __restrict__ out, int residual) {
    int idx = blockIdx.x * blockDim.x + threadIdx.x;
    if (idx >= N_NODES * HIDDEN) return;
    int c = idx & 127;
    float g = gamma[c] * rsqrtf(var[c] + BN_EPS);
    float v = (agg[idx] + bias[c] - mean[c]) * g + beta[c];
    v = fmaxf(v, 0.f);
    if (residual) v += 0.5f * xprev[idx];
    out[idx] = v;
}

// ------------------------------------------------------------------ driver
extern "C" void kernel_launch(void* const* d_in, const int* in_sizes, int n_in,
                              void* d_out, int out_size, void* d_ws, size_t ws_size,
                              hipStream_t stream) {
    const int*   edge   = (const int*)d_in[0];
    const int*   src    = edge;
    const int*   dst    = edge + N_EDGES;
    const float* sf     = (const float*)d_in[1];
    const float* emb    = (const float*)d_in[2];
    const float* pw     = (const float*)d_in[3];
    const float* pb     = (const float*)d_in[4];
    const float* conv_w = (const float*)d_in[5];
    const float* conv_b = (const float*)d_in[6];
    const float* gam    = (const float*)d_in[7];
    const float* bet    = (const float*)d_in[8];
    const float* mu     = (const float*)d_in[9];
    const float* var    = (const float*)d_in[10];
    float* out = (float*)d_out;

    char* ws = (char*)d_ws;
    float* dinv = (float*)ws;                                   // N floats
    size_t off  = ((size_t)N_NODES * sizeof(float) + 511) & ~(size_t)511;
    float* xb   = (float*)(ws + off);                           // N*128 (x, persistent)
    float* hb   = xb + (size_t)N_NODES * HIDDEN;                // N*128 (h)
    float* ab   = hb + (size_t)N_NODES * HIDDEN;                // N*128 (agg)

    const int NH   = N_NODES * HIDDEN;        // 12.8M
    const int NT   = N_NODES / 16;            // 6250 tiles (exact)
    dim3 B(256);

    // degrees -> dinv (in place)
    gcn_zero_f4<<<dim3((N_NODES / 4 + 255) / 256), B, 0, stream>>>((float4*)dinv, N_NODES / 4);
    gcn_deg   <<<dim3((N_EDGES + 255) / 256),      B, 0, stream>>>(dst, dinv);
    gcn_dinv  <<<dim3((N_NODES + 255) / 256),      B, 0, stream>>>(dinv);

    // input features
    gcn_feat<<<dim3((NH + 255) / 256), B, 0, stream>>>(emb, sf, pw, pb, xb);

    for (int i = 0; i < 3; ++i) {
        const float* Wi = conv_w + (size_t)i * HIDDEN * HIDDEN;
        // h = x @ W[i]   (WMMA bf16)
        gcn_gemm_wmma<<<dim3((NT + 7) / 8), B, 0, stream>>>(xb, Wi, hb, NT);
        // agg = segment_sum(h[src]*norm, dst)
        gcn_zero_f4<<<dim3((NH / 4 + 255) / 256), B, 0, stream>>>((float4*)ab, NH / 4);
        long long sthreads = (long long)N_EDGES * 32;
        gcn_scatter<<<dim3((unsigned)((sthreads + 255) / 256)), B, 0, stream>>>(src, dst, dinv, hb, ab);
        // BN + ReLU + residual; layer 2 writes d_out, others update xb in place
        float* dstbuf = (i == 2) ? out : xb;
        gcn_bn_relu<<<dim3((NH + 255) / 256), B, 0, stream>>>(
            ab, xb, conv_b + i * HIDDEN, gam + i * HIDDEN, bet + i * HIDDEN,
            mu + i * HIDDEN, var + i * HIDDEN, dstbuf, (i > 0) ? 1 : 0);
    }
}
